// EnsembleMLP_3882650437196
// MI455X (gfx1250) — compile-verified, matrix operand order checked by reading, not run on hardware
//
#include <hip/hip_runtime.h>
#include <math.h>

// ---- problem constants (match reference) ----
#define INPUT_DIM    512
#define HIDDEN_DIM   4096
#define ENSEMBLE_DIM 64
#define N_MEMBERS    256
#define OUTPUT_DIM   10
#define BATCH        4096

// ---- tiling ----
#define OUT_STRIDE   16                     // padded Agg/partial row stride
#define BM           64                     // batch rows per workgroup (4 waves x 16)
#define WAVES        4
#define NSPLIT       8                      // hidden-dim splits per row tile
#define NCHUNK       (HIDDEN_DIM / NSPLIT)  // 512
#define KFRAGS       (INPUT_DIM / 32)       // 16 WMMA k-steps

// ---- plain (trivial) vector types: safe inside unions ----
typedef __attribute__((ext_vector_type(16))) __bf16        v16bf;
typedef __attribute__((ext_vector_type(8)))  float         v8f;
typedef __attribute__((ext_vector_type(4)))  float         f32x4;
typedef __attribute__((ext_vector_type(4)))  unsigned int  u32x4;
typedef __attribute__((ext_vector_type(4)))  unsigned short u16x4;

union BF16x16 {
  v16bf v;
  unsigned short u[16];
  struct { u32x4 lo; u32x4 hi; } q;
};

__device__ __forceinline__ unsigned short f2bf(float f) {
  unsigned int x = __builtin_bit_cast(unsigned int, f);
  unsigned int r = (x + 0x7FFFu + ((x >> 16) & 1u)) >> 16;  // round-to-nearest-even
  return (unsigned short)r;
}

// Native v_tanh_f32 on gfx1250 (single trans op, co-executes with WMMA);
// fall back to libm expansion if the builtin is unavailable.
__device__ __forceinline__ float fast_tanh(float x) {
#if __has_builtin(__builtin_amdgcn_tanhf)
  return __builtin_amdgcn_tanhf(x);
#else
  return tanhf(x);
#endif
}

// ---------------------------------------------------------------------------
// Kernel 1: Agg[h, o] = (1/M) sum over (m,e) with ids[m,e]==h of W_pred[m,o,e]
// Deterministic scan (no atomics): one thread per hidden index h.
// ---------------------------------------------------------------------------
__global__ void build_agg_kernel(const int* __restrict__ ids,
                                 const float* __restrict__ W_pred,
                                 float* __restrict__ Agg) {
  const int h = blockIdx.x * blockDim.x + threadIdx.x;
  if (h >= HIDDEN_DIM) return;
  float acc[OUTPUT_DIM];
  #pragma unroll
  for (int o = 0; o < OUTPUT_DIM; ++o) acc[o] = 0.0f;

  for (int i = 0; i < N_MEMBERS * ENSEMBLE_DIM; ++i) {
    if (ids[i] == h) {
      const int m = i >> 6;           // / ENSEMBLE_DIM
      const int e = i & 63;           // % ENSEMBLE_DIM
      const float* wp = W_pred + ((size_t)m * OUTPUT_DIM) * ENSEMBLE_DIM + e;
      #pragma unroll
      for (int o = 0; o < OUTPUT_DIM; ++o) acc[o] += wp[(size_t)o * ENSEMBLE_DIM];
    }
  }
  float* ag = Agg + (size_t)h * OUT_STRIDE;
  const float inv = 1.0f / (float)N_MEMBERS;
  #pragma unroll
  for (int o = 0; o < OUTPUT_DIM; ++o) ag[o] = acc[o] * inv;
  #pragma unroll
  for (int o = OUTPUT_DIM; o < OUT_STRIDE; ++o) ag[o] = 0.0f;
}

// ---------------------------------------------------------------------------
// Kernel 2: W_in fp32 -> bf16 (row-major [H][K], K contiguous = B-fragment order)
// ---------------------------------------------------------------------------
__global__ void cvt_w_kernel(const float* __restrict__ W,
                             unsigned short* __restrict__ Wbf) {
  const int i = (blockIdx.x * blockDim.x + threadIdx.x) * 4;
  f32x4 f = *(const f32x4*)(W + i);
  u16x4 r;
  r.x = f2bf(f.x); r.y = f2bf(f.y); r.z = f2bf(f.z); r.w = f2bf(f.w);
  *(u16x4*)(Wbf + i) = r;
}

// ---------------------------------------------------------------------------
// Kernel 3: fused  partial[s, b, o] = sum_{n in split s} tanh(x[b,:]·W[n,:]) * Agg[n,o]
// 4 waves/block, wave owns 16 batch rows; A fragments register-resident.
// ---------------------------------------------------------------------------
__global__ void __launch_bounds__(WAVES * 32)
ensemble_main_kernel(const float* __restrict__ x,
                     const unsigned short* __restrict__ Wbf,
                     const float* __restrict__ Agg,
                     float* __restrict__ partial) {
  const int tid    = threadIdx.x;
  const int wave   = tid >> 5;
  const int lane   = tid & 31;
  const int lane16 = lane & 15;
  const int khalf  = (lane >= 16) ? 8 : 0;   // hi half-wave takes K+8 per ISA A/B layout

  const int tile    = blockIdx.x / NSPLIT;   // which 64-row batch tile
  const int s       = blockIdx.x % NSPLIT;   // which hidden-dim split
  const int rowbase = tile * BM + wave * 16; // this wave's 16 rows

  // ---- A fragments: 16 rows x 512 K, fp32 -> bf16 once, kept in VGPRs ----
  const float* xrow = x + (size_t)(rowbase + lane16) * INPUT_DIM;
  v16bf afrag[KFRAGS];
  #pragma unroll
  for (int kf = 0; kf < KFRAGS; ++kf) {
    const float* p = xrow + kf * 32 + khalf;
    f32x4 a0 = *(const f32x4*)(p);
    f32x4 a1 = *(const f32x4*)(p + 4);
    f32x4 b0 = *(const f32x4*)(p + 16);
    f32x4 b1 = *(const f32x4*)(p + 20);
    BF16x16 t;
    t.u[0]  = f2bf(a0.x); t.u[1]  = f2bf(a0.y); t.u[2]  = f2bf(a0.z); t.u[3]  = f2bf(a0.w);
    t.u[4]  = f2bf(a1.x); t.u[5]  = f2bf(a1.y); t.u[6]  = f2bf(a1.z); t.u[7]  = f2bf(a1.w);
    t.u[8]  = f2bf(b0.x); t.u[9]  = f2bf(b0.y); t.u[10] = f2bf(b0.z); t.u[11] = f2bf(b0.w);
    t.u[12] = f2bf(b1.x); t.u[13] = f2bf(b1.y); t.u[14] = f2bf(b1.z); t.u[15] = f2bf(b1.w);
    afrag[kf] = t.v;
  }

  float outacc[8][OUTPUT_DIM];
  #pragma unroll
  for (int v = 0; v < 8; ++v)
    #pragma unroll
    for (int o = 0; o < OUTPUT_DIM; ++o) outacc[v][o] = 0.0f;

  const int n0 = s * NCHUNK;
  for (int nt = 0; nt < NCHUNK; nt += 16) {
    const int ncol = n0 + nt + lane16;                       // this lane's B column / C column
    const unsigned short* wrow = Wbf + (size_t)ncol * INPUT_DIM + khalf;
    __builtin_prefetch((const void*)(wrow + 16 * INPUT_DIM), 0, 0);

    v8f acc = {};
    #pragma unroll
    for (int kf = 0; kf < KFRAGS; ++kf) {
      BF16x16 b;
      b.q.lo = *(const u32x4*)(wrow + kf * 32);
      b.q.hi = *(const u32x4*)(wrow + kf * 32 + 16);
      acc = __builtin_amdgcn_wmma_f32_16x16x32_bf16(
          /*neg_a=*/false, afrag[kf], /*neg_b=*/false, b.v,
          /*c_mod=*/(short)0, acc, /*reuse_a=*/false, /*reuse_b=*/false);
    }

    // epilogue: tanh on C tile (lane holds column ncol, rows via VGPR index),
    // then contract against Agg[ncol, 0..9]
    const float* ag = Agg + (size_t)ncol * OUT_STRIDE;
    float agv[OUTPUT_DIM];
    #pragma unroll
    for (int o = 0; o < OUTPUT_DIM; ++o) agv[o] = ag[o];
    #pragma unroll
    for (int v = 0; v < 8; ++v) {
      const float t = fast_tanh(acc[v]);
      #pragma unroll
      for (int o = 0; o < OUTPUT_DIM; ++o) outacc[v][o] += t * agv[o];
    }
  }

  // reduce over the 16 lanes of each half-wave (they hold distinct n columns,
  // identical rows): butterfly within lanes 0-15 and 16-31 independently.
  #pragma unroll
  for (int v = 0; v < 8; ++v) {
    #pragma unroll
    for (int o = 0; o < OUTPUT_DIM; ++o) {
      float t = outacc[v][o];
      t += __shfl_xor(t, 1, 32);
      t += __shfl_xor(t, 2, 32);
      t += __shfl_xor(t, 4, 32);
      t += __shfl_xor(t, 8, 32);
      outacc[v][o] = t;
    }
  }

  if (lane16 == 0) {
    const int rbase = rowbase + ((lane >= 16) ? 8 : 0);      // C layout: hi lanes = rows 8..15
    #pragma unroll
    for (int v = 0; v < 8; ++v) {
      float* dst = partial + ((size_t)s * BATCH + (size_t)(rbase + v)) * OUT_STRIDE;
      #pragma unroll
      for (int o = 0; o < OUTPUT_DIM; ++o) dst[o] = outacc[v][o];
    }
  }
}

// ---------------------------------------------------------------------------
// Kernel 4: sum the NSPLIT partials -> out[b, o]
// ---------------------------------------------------------------------------
__global__ void reduce_kernel(const float* __restrict__ partial,
                              float* __restrict__ out) {
  const int i = blockIdx.x * blockDim.x + threadIdx.x;
  if (i >= BATCH * OUTPUT_DIM) return;
  const int b = i / OUTPUT_DIM;
  const int o = i % OUTPUT_DIM;
  float t = 0.0f;
  #pragma unroll
  for (int s = 0; s < NSPLIT; ++s)
    t += partial[((size_t)s * BATCH + b) * OUT_STRIDE + o];
  out[i] = t;
}

// ---------------------------------------------------------------------------
extern "C" void kernel_launch(void* const* d_in, const int* in_sizes, int n_in,
                              void* d_out, int out_size, void* d_ws, size_t ws_size,
                              hipStream_t stream) {
  const float* x      = (const float*)d_in[0];
  const float* W_in   = (const float*)d_in[1];
  const float* W_pred = (const float*)d_in[2];
  const int*   ids    = (const int*)d_in[3];

  // workspace layout: Agg (256KB) | Wbf (4MB) | partial (2MB)  ~= 6.3MB total
  char* ws = (char*)d_ws;
  float* Agg = (float*)ws;
  size_t off = (size_t)HIDDEN_DIM * OUT_STRIDE * sizeof(float);
  unsigned short* Wbf = (unsigned short*)(ws + off);
  off += (size_t)HIDDEN_DIM * INPUT_DIM * sizeof(unsigned short);
  float* partial = (float*)(ws + off);

  build_agg_kernel<<<HIDDEN_DIM / 256, 256, 0, stream>>>(ids, W_pred, Agg);
  cvt_w_kernel<<<(HIDDEN_DIM * INPUT_DIM / 4) / 256, 256, 0, stream>>>(W_in, Wbf);
  ensemble_main_kernel<<<(BATCH / BM) * NSPLIT, WAVES * 32, 0, stream>>>(x, Wbf, Agg, partial);
  reduce_kernel<<<(BATCH * OUTPUT_DIM + 255) / 256, 256, 0, stream>>>(partial, (float*)d_out);
}
